// NSAAttention_78194174591050
// MI455X (gfx1250) — matrix operand order repaired and checked by reading, not compiled, not run on hardware
//
#include <hip/hip_runtime.h>
#include <hip/hip_bf16.h>

typedef __attribute__((ext_vector_type(16))) _Float16 v16h;
typedef __attribute__((ext_vector_type(8)))  _Float16 h8;
typedef __attribute__((ext_vector_type(4)))  _Float16 h4;
typedef __attribute__((ext_vector_type(8)))  float    v8f;
typedef __attribute__((ext_vector_type(4)))  float    f4;

#define Zc 4
#define Hc 16
#define Gc 4
#define Sc 4096
#define NBc 128
#define Dc 128
#define SCALE 0.08838834764831845f   // 1/sqrt(128)

// LDS layout (dynamic): K f16 [NB][D] (32KB) | V^T f16 [D][NB] (32KB) | P f16 per-wave [16][D] (8*4KB)
#define SMEM_BYTES ((size_t)(NBc*Dc + Dc*NBc + 8*16*Dc) * sizeof(_Float16))  // 98304

// Butterfly reductions across the 16 lanes of a half-wave using ds_swizzle_b32
// group-of-32 XOR mode: offset = and_mask(0x1F) | xor_mask<<10. XOR of 1,2,4,8
// never crosses bit4, so each 16-lane half reduces independently.
__device__ __forceinline__ float swz_xor(float x, const int imm_off) {
    // imm_off must be a literal at each call site (builtin takes an ICE).
    return x; // overridden below by macro
}
#define SWZ(x, imm) __int_as_float(__builtin_amdgcn_ds_swizzle(__float_as_int(x), (imm)))

__device__ __forceinline__ float red_max16(float x) {
    x = fmaxf(x, SWZ(x, 0x041F));   // xor 1
    x = fmaxf(x, SWZ(x, 0x081F));   // xor 2
    x = fmaxf(x, SWZ(x, 0x101F));   // xor 4
    x = fmaxf(x, SWZ(x, 0x201F));   // xor 8
    return x;
}
__device__ __forceinline__ float red_add16(float x) {
    x += SWZ(x, 0x041F);
    x += SWZ(x, 0x081F);
    x += SWZ(x, 0x101F);
    x += SWZ(x, 0x201F);
    return x;
}

__global__ void __launch_bounds__(256)
nsa_cmp_attn_fwd(const float* __restrict__ q,
                 const float* __restrict__ kb,
                 const float* __restrict__ vb,
                 const int*   __restrict__ bes,
                 float*       __restrict__ out)
{
    extern __shared__ __align__(32) char smem_raw[];
    _Float16* ks  = (_Float16*)smem_raw;          // K  [blk][dim]
    _Float16* vts = ks  + NBc * Dc;               // Vt [dim][blk]
    _Float16* pba = vts + Dc * NBc;               // P  per-wave [16][dim=blk]

    const int tid    = threadIdx.x;
    const int lane   = tid & 31;
    const int wave   = tid >> 5;
    const int z      = blockIdx.z;
    const int h      = blockIdx.y;
    const int g      = h / (Hc / Gc);
    const int row0   = blockIdx.x * 128 + wave * 16;  // first query row for this wave
    const int half16 = lane >> 4;                     // which 16-lane half
    const int lcol   = lane & 15;
    const int h8off  = half16 * 8;

    // ---- Stage K (f16) and V^T (f16) into LDS: 2 * 16384 f32 elements, 256 threads ----
    {
        const f4* kb4 = (const f4*)(kb + (size_t)(z * Gc + g) * NBc * Dc);
        const f4* vb4 = (const f4*)(vb + (size_t)(z * Gc + g) * NBc * Dc);
        #pragma unroll
        for (int it = 0; it < 16; ++it) {
            int gi = it * 256 + tid;     // float4 index
            int r  = gi >> 5;            // block row (32 float4 per 128-wide row)
            int c4 = gi & 31;
            f4 xk = kb4[gi];
            h4 hk;
            hk[0] = (_Float16)xk[0]; hk[1] = (_Float16)xk[1];
            hk[2] = (_Float16)xk[2]; hk[3] = (_Float16)xk[3];
            *(h4*)(ks + r * Dc + c4 * 4) = hk;
            f4 xv = vb4[gi];
            vts[(c4 * 4 + 0) * NBc + r] = (_Float16)xv[0];
            vts[(c4 * 4 + 1) * NBc + r] = (_Float16)xv[1];
            vts[(c4 * 4 + 2) * NBc + r] = (_Float16)xv[2];
            vts[(c4 * 4 + 3) * NBc + r] = (_Float16)xv[3];
        }
    }
    __syncthreads();

    // block_ends for the 8 tile-columns this lane touches in the C/D layout
    int bec[8];
    #pragma unroll
    for (int t = 0; t < 8; ++t) bec[t] = bes[t * 16 + lcol];

    // ---- Q A-fragments straight from global (A layout: lane%16 = row M) ----
    const float* qrow = q + ((size_t)(z * Hc + h) * Sc + (row0 + lcol)) * Dc;
    v16h afrag[4];
    #pragma unroll
    for (int c = 0; c < 4; ++c) {
        const f4* p0 = (const f4*)(qrow + c * 32 + half16 * 16);      // K = c*32 + half16*16 + 0..7
        const f4* p1 = (const f4*)(qrow + c * 32 + half16 * 16 + 8);  // K = c*32 + half16*16 + 8..15
        f4 a0 = p0[0], a1 = p0[1], b0 = p1[0], b1 = p1[1];
        v16h a;
        #pragma unroll
        for (int j = 0; j < 4; ++j) {
            a[j]      = (_Float16)(a0[j] * SCALE);
            a[4 + j]  = (_Float16)(a1[j] * SCALE);
            a[8 + j]  = (_Float16)(b0[j] * SCALE);
            a[12 + j] = (_Float16)(b1[j] * SCALE);
        }
        afrag[c] = a;
    }

    // ---- S = Q * K^T : 8 N-tiles (16 blocks each) x 4 K-chunks of 32 ----
    v8f sacc[8];
    #pragma unroll
    for (int t = 0; t < 8; ++t) {
        v8f c = {};
        #pragma unroll
        for (int kc = 0; kc < 4; ++kc) {
            const v16h* bp = (const v16h*)(ks + (t * 16 + lcol) * Dc + kc * 32 + half16 * 16);
            c = __builtin_amdgcn_wmma_f32_16x16x32_f16(false, afrag[kc], false, *bp,
                                                       (short)0, c, false, false);
        }
        sacc[t] = c;
    }

    // ---- masked softmax in registers (row = half16*8 + i; cols across 16 lanes x 8 tiles) ----
    float rinv[8];
    #pragma unroll
    for (int i = 0; i < 8; ++i) {
        const int mpos = row0 + half16 * 8 + i;        // query position for this row
        float mx = -3.0e38f;
        #pragma unroll
        for (int t = 0; t < 8; ++t)
            if (bec[t] <= mpos) mx = fmaxf(mx, sacc[t][i]);
        mx = red_max16(mx);                            // row max across 16 lanes
        float sum = 0.0f;
        #pragma unroll
        for (int t = 0; t < 8; ++t) {
            float p = (bec[t] <= mpos) ? __expf(sacc[t][i] - mx) : 0.0f;
            sacc[t][i] = p;
            sum += p;
        }
        sum = red_add16(sum);                          // row denom across 16 lanes
        // l >= 1 when any block is valid (max term contributes exp(0)=1), so
        // a single-cycle v_rcp_f32 is safe; l == 0 rows output zeros.
        rinv[i] = (sum > 0.0f) ? __builtin_amdgcn_rcpf(sum) : 0.0f;
    }

    // ---- write P (f16) to per-wave LDS, re-layout C/D -> A ----
    _Float16* pb = pba + wave * (16 * Dc);
    #pragma unroll
    for (int t = 0; t < 8; ++t) {
        #pragma unroll
        for (int i = 0; i < 8; ++i)
            pb[(half16 * 8 + i) * Dc + t * 16 + lcol] = (_Float16)sacc[t][i];
    }
    asm volatile("s_wait_dscnt 0" ::: "memory");        // wave-local LDS visibility

    // ---- P A-fragments from LDS ----
    v16h pfrag[4];
    #pragma unroll
    for (int c = 0; c < 4; ++c) {
        const h8* p0 = (const h8*)(pb + lcol * Dc + c * 32 + half16 * 16);
        const h8* p1 = (const h8*)(pb + lcol * Dc + c * 32 + half16 * 16 + 8);
        h8 lo = *p0, hi = *p1;
        v16h a;
        #pragma unroll
        for (int j = 0; j < 8; ++j) { a[j] = lo[j]; a[8 + j] = hi[j]; }
        pfrag[c] = a;
    }

    // ---- O = P * V (B from transposed-V LDS image), scale by 1/l, store fp32 ----
    float* orow = out + ((size_t)(z * Hc + h) * Sc + row0) * Dc;
    #pragma unroll
    for (int t = 0; t < 8; ++t) {
        v8f o = {};
        #pragma unroll
        for (int kc = 0; kc < 4; ++kc) {
            const v16h* bp = (const v16h*)(vts + (t * 16 + lcol) * NBc + kc * 32 + half16 * 16);
            o = __builtin_amdgcn_wmma_f32_16x16x32_f16(false, pfrag[kc], false, *bp,
                                                       (short)0, o, false, false);
        }
        #pragma unroll
        for (int i = 0; i < 8; ++i)
            orow[(half16 * 8 + i) * Dc + t * 16 + lcol] = o[i] * rinv[i];
    }
}

extern "C" void kernel_launch(void* const* d_in, const int* in_sizes, int n_in,
                              void* d_out, int out_size, void* d_ws, size_t ws_size,
                              hipStream_t stream) {
    const float* q   = (const float*)d_in[0];
    const float* kb  = (const float*)d_in[1];
    const float* vb  = (const float*)d_in[2];
    const int*   bes = (const int*)d_in[3];
    float*       out = (float*)d_out;

    dim3 grid(Sc / 128, Hc, Zc);   // 32 x 16 x 4 = 2048 workgroups
    dim3 block(256);               // 8 waves (wave32), 16 query rows each
    nsa_cmp_attn_fwd<<<grid, block, SMEM_BYTES, stream>>>(q, kb, vb, bes, out);
}
